// HyperGCN_9749575762795
// MI455X (gfx1250) — matrix operand order repaired and checked by reading, not compiled, not run on hardware
//
#include <hip/hip_runtime.h>
#include <hip/hip_bf16.h>

typedef __attribute__((ext_vector_type(16))) _Float16 v16h;
typedef __attribute__((ext_vector_type(8)))  float    v8f;

#define NNODES 32768
#define NEDGES 4096
#define NINC   131072
#define DMODEL 256
#define NB     4
#define NC     64
#define EPSF   1e-5f

__device__ __forceinline__ float leakyf(float v) { return v >= 0.f ? v : 0.2f * v; }

__device__ __forceinline__ void atomAddF(float* p, float v) {
    __hip_atomic_fetch_add(p, v, __ATOMIC_RELAXED, __HIP_MEMORY_SCOPE_AGENT);
}

// float atomic max via ordered-int trick (works for mixed signs, -inf init)
__device__ __forceinline__ void atomMaxF(float* p, float v) {
    if (v >= 0.f) atomicMax((int*)p, __float_as_int(v));
    else          atomicMin((unsigned int*)p, __float_as_uint(v));
}

// ---------------- small prep kernels ----------------

__global__ void k_f32_to_f16(const float* __restrict__ src, _Float16* __restrict__ dst, int n) {
    int i = blockIdx.x * blockDim.x + threadIdx.x;
    if (i < n) dst[i] = (_Float16)src[i];
}

// Pack W (Dout x K=256, row-major f32) into WMMA-fragment-ordered fp16:
// Bp flat index = ((((cb*8 + kc)*8 + ct)*32 + lane)*16 + e)
//   col d = cb*128 + ct*16 + (lane&15) ;  k = kc*32 + (lane&16) + e
// so each lane's 16 B-matrix elements (K=khalf..khalf+15 of its column) are contiguous.
__global__ void k_pack_B(const float* __restrict__ W, _Float16* __restrict__ Bp,
                         int Dout, int K) {
    int i = blockIdx.x * blockDim.x + threadIdx.x;
    if (i >= Dout * K) return;
    int e    = i & 15;
    int lane = (i >> 4) & 31;
    int ct   = (i >> 9) & 7;
    int kc   = (i >> 12) & 7;          // K = 256 -> 8 chunks of 32
    int cb   = i >> 15;
    int d = cb * 128 + ct * 16 + (lane & 15);
    int k = kc * 32 + (lane & 16) + e;
    Bp[i] = (_Float16)W[(size_t)d * K + k];
}

__global__ void k_fillf(float* __restrict__ p, float v, int n) {
    int i = blockIdx.x * blockDim.x + threadIdx.x;
    if (i < n) p[i] = v;
}

// att: (2, 512)  -> attn_v[h*256+d]=att[h*512+d], atte_v[h*256+d]=att[h*512+256+d]
__global__ void k_prep_att(const float* __restrict__ att,
                           float* __restrict__ attn_v, float* __restrict__ atte_v) {
    int t = threadIdx.x;                       // 512 threads
    int h = t >> 8, d = t & 255;
    attn_v[t] = att[h * 512 + d];
    atte_v[t] = att[h * 512 + 256 + d];
}

// ---------------- WMMA tiled GEMM: C(M x Ncols) = A(M x 256) * B ----------------
// block: 256 thr = 8 waves; block tile = 32 rows x 128 cols; wave = 2 stacked 16x16 tiles.
// B comes pre-packed in fragment order (k_pack_B) -> one v16h global load per K-step,
// reused by two WMMAs. A staged once in LDS (32x256 f16 = 16KB), single barrier.
// mode 0: lin1  -> v=bn1(leaky(v+bias));  store f32 + f16
// mode 1: plain -> store f32
// mode 2: lin2  -> v=resid+leaky(v+bias); store f32
__global__ __launch_bounds__(256) void k_wmma_gemm(
    const _Float16* __restrict__ A, const _Float16* __restrict__ Bp,
    float* __restrict__ Cout, _Float16* __restrict__ CoutH,
    const float* __restrict__ bias,
    const float* __restrict__ bn_g, const float* __restrict__ bn_b,
    const float* __restrict__ bn_m, const float* __restrict__ bn_v,
    const float* __restrict__ resid,
    int Ncols, int mode)
{
    __shared__ __align__(16) _Float16 sA[32 * 256];

    const int tid  = threadIdx.x;
    const int lane = tid & 31;
    const int wave = tid >> 5;                 // 0..7 -> 16-col subtile
    const int row0 = blockIdx.x * 32;
    const int cb   = blockIdx.y;               // 128-col block

    // stage A tile: 32x256 halves = 1024 uint4, 4 per thread
    {
        const uint4* src = (const uint4*)(A + (size_t)row0 * 256);
        uint4* dst = (uint4*)sA;
        #pragma unroll
        for (int i = 0; i < 4; ++i) dst[tid + i * 256] = src[tid + i * 256];
    }
    __syncthreads();

    v8f acc0 = {}, acc1 = {};
    const _Float16* bptr = Bp + ((((size_t)cb * 8) * 8 + wave) * 32 + lane) * 16;
    const int m     = lane & 15;
    const int abase = (lane < 16) ? 0 : 16;

    #pragma unroll
    for (int kc = 0; kc < 8; ++kc) {
        const int k0 = kc * 32;
        v16h bf = *(const v16h*)(bptr + (size_t)kc * 4096);   // 8 waves*32 lanes*16
        v16h a0, a1;
        #pragma unroll
        for (int e = 0; e < 8; ++e) {
            a0[e]     = sA[m * 256        + k0 + abase + e];
            a0[8 + e] = sA[m * 256        + k0 + 8 + abase + e];
            a1[e]     = sA[(m + 16) * 256 + k0 + abase + e];
            a1[8 + e] = sA[(m + 16) * 256 + k0 + 8 + abase + e];
        }
        acc0 = __builtin_amdgcn_wmma_f32_16x16x32_f16(false, a0, false, bf,
                                                      (short)0, acc0, false, false);
        acc1 = __builtin_amdgcn_wmma_f32_16x16x32_f16(false, a1, false, bf,
                                                      (short)0, acc1, false, false);
    }

    // epilogue: C/D layout: lanes 0-15: M=i, N=lane; lanes 16-31: M=8+i, N=lane-16
    const int ncol  = cb * 128 + wave * 16 + (lane & 15);
    const int mbase = (lane < 16) ? 0 : 8;
    #pragma unroll
    for (int t = 0; t < 2; ++t) {
        v8f acc = t ? acc1 : acc0;
        #pragma unroll
        for (int i = 0; i < 8; ++i) {
            const int r = row0 + t * 16 + mbase + i;
            const size_t off = (size_t)r * Ncols + ncol;
            float v = acc[i];
            if (mode == 0) {
                v = leakyf(v + bias[ncol]);
                v = bn_g[ncol] * (v - bn_m[ncol]) * rsqrtf(bn_v[ncol] + EPSF) + bn_b[ncol];
                Cout[off]  = v;
                CoutH[off] = (_Float16)v;
            } else if (mode == 1) {
                Cout[off] = v;
            } else {
                v = leakyf(v + bias[ncol]);
                Cout[off] = resid[off] + v;
            }
        }
    }
}

// ---------------- sparse / incidence kernels ----------------

// eattr[e,:] += h[n,:]   (one block per incidence, 256 dims)
__global__ __launch_bounds__(256) void k_edge_gather(
    const int* __restrict__ idx_n, const int* __restrict__ idx_e,
    const float* __restrict__ h, float* __restrict__ eattr)
{
    int i = blockIdx.x, t = threadIdx.x;
    int n = idx_n[i], e = idx_e[i];
    atomAddF(&eattr[(size_t)e * 256 + t], h[(size_t)n * 256 + t]);
}

// row-dot with attention vector: out[row,h] = sum_d X[row, h*256+d]*attv[h*256+d]
__global__ __launch_bounds__(512) void k_scores(
    const float* __restrict__ X, const float* __restrict__ attv,
    float* __restrict__ out)
{
    __shared__ float s[512];
    int row = blockIdx.x, t = threadIdx.x;
    s[t] = X[(size_t)row * 512 + t] * attv[t];
    __syncthreads();
    for (int st = 128; st > 0; st >>= 1) {
        if ((t & 255) < st) s[t] += s[t + st];
        __syncthreads();
    }
    if (t == 0)   out[row * 2 + 0] = s[0];
    if (t == 256) out[row * 2 + 1] = s[256];
}

// alpha_raw = leaky(sn[n]+se[e]); segment max into amax[n]; degrees
__global__ void k_alpha1(const int* __restrict__ idx_n, const int* __restrict__ idx_e,
                         const float* __restrict__ sn, const float* __restrict__ se,
                         float* __restrict__ alpha, float* __restrict__ amax,
                         float* __restrict__ degn, float* __restrict__ dege)
{
    int i = blockIdx.x * blockDim.x + threadIdx.x;
    if (i >= NINC) return;
    int n = idx_n[i], e = idx_e[i];
    #pragma unroll
    for (int hh = 0; hh < 2; ++hh) {
        float a = leakyf(sn[n * 2 + hh] + se[e * 2 + hh]);
        alpha[i * 2 + hh] = a;
        atomMaxF(&amax[n * 2 + hh], a);
    }
    atomAddF(&degn[n], 1.0f);
    atomAddF(&dege[e], 1.0f);
}

// alpha := exp(alpha - amax_fin[n]); denom[n] += alpha
__global__ void k_alpha2(const int* __restrict__ idx_n,
                         const float* __restrict__ amax,
                         float* __restrict__ alpha, float* __restrict__ denom)
{
    int i = blockIdx.x * blockDim.x + threadIdx.x;
    if (i >= NINC) return;
    int n = idx_n[i];
    #pragma unroll
    for (int hh = 0; hh < 2; ++hh) {
        float am = amax[n * 2 + hh];
        if (!isfinite(am)) am = 0.f;
        float ex = __expf(alpha[i * 2 + hh] - am);
        alpha[i * 2 + hh] = ex;
        atomAddF(&denom[n * 2 + hh], ex);
    }
}

// m1: edge_out[e,h,:] += Binv[e]*alpha*xh[n,h,:]; also store normalized alpha
__global__ __launch_bounds__(256) void k_m1(
    const int* __restrict__ idx_n, const int* __restrict__ idx_e,
    const float* __restrict__ ex, const float* __restrict__ denom,
    const float* __restrict__ dege, const float* __restrict__ xh,
    float* __restrict__ edge_out, float* __restrict__ alpha_fin)
{
    int i = blockIdx.x, t = threadIdx.x;
    int n = idx_n[i], e = idx_e[i];
    float de   = dege[e];
    float binv = de > 0.f ? 1.f / de : 0.f;
    float a0 = ex[i * 2 + 0] / (denom[n * 2 + 0] + 1e-16f);
    float a1 = ex[i * 2 + 1] / (denom[n * 2 + 1] + 1e-16f);
    if (t == 0) { alpha_fin[i * 2 + 0] = a0; alpha_fin[i * 2 + 1] = a1; }
    atomAddF(&edge_out[(size_t)e * 512 + t],       binv * a0 * xh[(size_t)n * 512 + t]);
    atomAddF(&edge_out[(size_t)e * 512 + 256 + t], binv * a1 * xh[(size_t)n * 512 + 256 + t]);
}

// m2 + head-mean: x2[n,d] += 0.5*Dinv[n]*(a0*edge_out[e,0,d] + a1*edge_out[e,1,d])
__global__ __launch_bounds__(256) void k_m2(
    const int* __restrict__ idx_n, const int* __restrict__ idx_e,
    const float* __restrict__ alpha_fin, const float* __restrict__ degn,
    const float* __restrict__ edge_out, float* __restrict__ x2)
{
    int i = blockIdx.x, t = threadIdx.x;
    int n = idx_n[i], e = idx_e[i];
    float dn   = degn[n];
    float dinv = dn > 0.f ? 1.f / dn : 0.f;
    float a0 = alpha_fin[i * 2 + 0], a1 = alpha_fin[i * 2 + 1];
    float v = 0.5f * dinv * (a0 * edge_out[(size_t)e * 512 + t] +
                             a1 * edge_out[(size_t)e * 512 + 256 + t]);
    atomAddF(&x2[(size_t)n * 256 + t], v);
}

// t = bn2(h + x2 + cb)  -> fp16 A for final GEMM
__global__ void k_prep2(const float* __restrict__ h, const float* __restrict__ x2,
                        const float* __restrict__ cb,
                        const float* __restrict__ g, const float* __restrict__ b,
                        const float* __restrict__ m, const float* __restrict__ var,
                        _Float16* __restrict__ t16)
{
    int i = blockIdx.x * blockDim.x + threadIdx.x;
    if (i >= NNODES * 256) return;
    int d = i & 255;
    float v = h[i] + x2[i] + cb[d];
    v = g[d] * (v - m[d]) * rsqrtf(var[d] + EPSF) + b[d];
    t16[i] = (_Float16)v;
}

// final LayerNorm over C=64 groups of the flat (B,N,C) tensor
__global__ __launch_bounds__(64) void k_ln(const float* __restrict__ y,
                                           const float* __restrict__ g,
                                           const float* __restrict__ b,
                                           float* __restrict__ out)
{
    __shared__ float s[64];
    int row = blockIdx.x, t = threadIdx.x;
    float v = y[(size_t)row * 64 + t];
    s[t] = v; __syncthreads();
    for (int st = 32; st > 0; st >>= 1) { if (t < st) s[t] += s[t + st]; __syncthreads(); }
    float mu = s[0] * (1.f / 64.f); __syncthreads();
    float d = v - mu;
    s[t] = d * d; __syncthreads();
    for (int st = 32; st > 0; st >>= 1) { if (t < st) s[t] += s[t + st]; __syncthreads(); }
    float var = s[0] * (1.f / 64.f);
    out[(size_t)row * 64 + t] = g[t] * d * rsqrtf(var + EPSF) + b[t];
}

// ---------------- host launch ----------------

extern "C" void kernel_launch(void* const* d_in, const int* in_sizes, int n_in,
                              void* d_out, int out_size, void* d_ws, size_t ws_size,
                              hipStream_t stream) {
    const float* x     = (const float*)d_in[0];
    const int*   he    = (const int*)  d_in[1];
    const float* W1    = (const float*)d_in[2];
    const float* b1    = (const float*)d_in[3];
    const float* bn1g  = (const float*)d_in[4];
    const float* bn1b  = (const float*)d_in[5];
    const float* bn1m  = (const float*)d_in[6];
    const float* bn1v  = (const float*)d_in[7];
    const float* attW  = (const float*)d_in[8];
    const float* att   = (const float*)d_in[9];
    const float* cb    = (const float*)d_in[10];
    const float* bn2g  = (const float*)d_in[11];
    const float* bn2b  = (const float*)d_in[12];
    const float* bn2m  = (const float*)d_in[13];
    const float* bn2v  = (const float*)d_in[14];
    const float* W2    = (const float*)d_in[15];
    const float* b2    = (const float*)d_in[16];
    const float* lng   = (const float*)d_in[17];
    const float* lnb   = (const float*)d_in[18];
    (void)in_sizes; (void)n_in; (void)out_size; (void)ws_size;

    const int* idx_n = he;
    const int* idx_e = he + NINC;

    // workspace carve-up
    char*  base = (char*)d_ws;
    size_t off  = 0;
    auto alloc = [&](size_t bytes) -> char* {
        char* p = base + off;
        off = (off + bytes + 255) & ~(size_t)255;
        return p;
    };
    float*     h_f    = (float*)    alloc((size_t)NNODES * 256 * 4);
    _Float16*  h_h    = (_Float16*) alloc((size_t)NNODES * 256 * 2);
    _Float16*  x_h    = (_Float16*) alloc((size_t)NNODES * 256 * 2);
    float*     eattr  = (float*)    alloc((size_t)NEDGES * 256 * 4);
    _Float16*  eattrh = (_Float16*) alloc((size_t)NEDGES * 256 * 2);
    float*     xh     = (float*)    alloc((size_t)NNODES * 512 * 4);
    float*     ea     = (float*)    alloc((size_t)NEDGES * 512 * 4);
    float*     sn     = (float*)    alloc((size_t)NNODES * 2 * 4);
    float*     se     = (float*)    alloc((size_t)NEDGES * 2 * 4);
    float*     amax   = (float*)    alloc((size_t)NNODES * 2 * 4);
    float*     denom  = (float*)    alloc((size_t)NNODES * 2 * 4);
    float*     degn   = (float*)    alloc((size_t)NNODES * 4);
    float*     dege   = (float*)    alloc((size_t)NEDGES * 4);
    float*     alpha  = (float*)    alloc((size_t)NINC * 2 * 4);
    float*     alfin  = (float*)    alloc((size_t)NINC * 2 * 4);
    float*     eout   = (float*)    alloc((size_t)NEDGES * 512 * 4);
    float*     x2     = (float*)    alloc((size_t)NNODES * 256 * 4);
    _Float16*  t_h    = (_Float16*) alloc((size_t)NNODES * 256 * 2);
    float*     ypre   = (float*)    alloc((size_t)NNODES * 256 * 4);
    float*     attnv  = (float*)    alloc(512 * 4);
    float*     attev  = (float*)    alloc(512 * 4);
    _Float16*  W1p    = (_Float16*) alloc((size_t)256 * 256 * 2);
    _Float16*  attWp  = (_Float16*) alloc((size_t)256 * 512 * 2);
    _Float16*  W2p    = (_Float16*) alloc((size_t)256 * 256 * 2);

    // zero accumulators (graph-capture safe)
    hipMemsetAsync(eattr, 0, (size_t)NEDGES * 256 * 4, stream);
    hipMemsetAsync(denom, 0, (size_t)NNODES * 2 * 4, stream);
    hipMemsetAsync(degn,  0, (size_t)NNODES * 4, stream);
    hipMemsetAsync(dege,  0, (size_t)NEDGES * 4, stream);
    hipMemsetAsync(eout,  0, (size_t)NEDGES * 512 * 4, stream);
    hipMemsetAsync(x2,    0, (size_t)NNODES * 256 * 4, stream);
    k_fillf<<<(NNODES * 2 + 255) / 256, 256, 0, stream>>>(amax, -INFINITY, NNODES * 2);

    // weights -> fragment-packed fp16, x -> fp16
    k_pack_B<<<(256 * 256 + 255) / 256, 256, 0, stream>>>(W1,   W1p,   256, 256);
    k_pack_B<<<(512 * 256 + 255) / 256, 256, 0, stream>>>(attW, attWp, 512, 256);
    k_pack_B<<<(256 * 256 + 255) / 256, 256, 0, stream>>>(W2,   W2p,   256, 256);
    k_f32_to_f16<<<(NNODES * 256 + 255) / 256, 256, 0, stream>>>(x, x_h, NNODES * 256);
    k_prep_att<<<1, 512, 0, stream>>>(att, attnv, attev);

    // lin1: h = bn1(leaky(x@W1^T + b1))
    k_wmma_gemm<<<dim3(NNODES / 32, 256 / 128), 256, 0, stream>>>(
        x_h, W1p, h_f, h_h, b1, bn1g, bn1b, bn1m, bn1v, nullptr, 256, 0);

    // eattr = segment_sum(h[idx_n] by idx_e)
    k_edge_gather<<<NINC, 256, 0, stream>>>(idx_n, idx_e, h_f, eattr);
    k_f32_to_f16<<<(NEDGES * 256 + 255) / 256, 256, 0, stream>>>(eattr, eattrh, NEDGES * 256);

    // xh = h@attW^T ; ea = eattr@attW^T
    k_wmma_gemm<<<dim3(NNODES / 32, 512 / 128), 256, 0, stream>>>(
        h_h, attWp, xh, nullptr, nullptr, nullptr, nullptr, nullptr, nullptr, nullptr, 512, 1);
    k_wmma_gemm<<<dim3(NEDGES / 32, 512 / 128), 256, 0, stream>>>(
        eattrh, attWp, ea, nullptr, nullptr, nullptr, nullptr, nullptr, nullptr, nullptr, 512, 1);

    // attention scores
    k_scores<<<NNODES, 512, 0, stream>>>(xh, attnv, sn);
    k_scores<<<NEDGES, 512, 0, stream>>>(ea, attev, se);

    // segment softmax over idx_n
    k_alpha1<<<NINC / 256, 256, 0, stream>>>(idx_n, idx_e, sn, se, alpha, amax, degn, dege);
    k_alpha2<<<NINC / 256, 256, 0, stream>>>(idx_n, amax, alpha, denom);

    // two-hop message passing
    k_m1<<<NINC, 256, 0, stream>>>(idx_n, idx_e, alpha, denom, dege, xh, eout, alfin);
    k_m2<<<NINC, 256, 0, stream>>>(idx_n, idx_e, alfin, degn, eout, x2);

    // lin2 + residual, then LayerNorm
    k_prep2<<<(NNODES * 256 + 255) / 256, 256, 0, stream>>>(
        h_f, x2, cb, bn2g, bn2b, bn2m, bn2v, t_h);
    k_wmma_gemm<<<dim3(NNODES / 32, 256 / 128), 256, 0, stream>>>(
        t_h, W2p, ypre, nullptr, b2, nullptr, nullptr, nullptr, nullptr, x, 256, 2);
    k_ln<<<NB * NNODES, 64, 0, stream>>>(ypre, lng, lnb, (float*)d_out);
}
